// GATmodel_33105607918117
// MI455X (gfx1250) — matrix operand order repaired and checked by reading, not compiled, not run on hardware
//
#include <hip/hip_runtime.h>

// ---------------- static problem sizes (match reference) ----------------
static constexpr int BG    = 512;          // graphs
static constexpr int NPER  = 64;           // nodes per graph
static constexpr int NN    = BG * NPER;    // 32768 nodes
static constexpr int EG    = BG * 256;     // 131072 edges
static constexpr int ET    = EG + NN;      // edges + self loops = 163840
static constexpr int FIN   = 37;
static constexpr int DM    = 128;
static constexpr int HG    = 8;            // GAT heads
static constexpr int HT    = 8;            // transformer heads
static constexpr int DH    = 16;           // DM / HT
static constexpr int SEEDS = 75;

typedef __attribute__((ext_vector_type(16))) _Float16 v16h;
typedef __attribute__((ext_vector_type(8)))  float    v8f;

#define DEVFN static __device__ __forceinline__

// ---- CDNA5 async global->LDS copy (guarded; falls back to plain LDS stores) ----
#if __has_builtin(__builtin_amdgcn_global_load_async_to_lds_b128) && \
    __has_builtin(__builtin_amdgcn_s_wait_asynccnt)
#define HAVE_ASYNC_LDS 1
typedef int v4i_ __attribute__((ext_vector_type(4)));
typedef __attribute__((address_space(1))) v4i_ g_v4i;   // global int4
typedef __attribute__((address_space(3))) v4i_ l_v4i;   // LDS int4
#else
#define HAVE_ASYNC_LDS 0
#endif

// ================= WMMA GEMM with LDS-staged B tile =================
// C[M,Nout] = act(A[M,K] @ W + bias). W pre-transposed+f16: WT[Nout][K].
// Preconditions: M%16==0, K%32==0 (K<=128), Nout%64==0.
// Block = 8 waves: 128 rows x 64 cols. B tile (64 rows of WT, contiguous 64*K
// halfs) is staged to LDS once per block (async if available), then each wave
// runs the K loop reading B fragments from LDS via ds_load.
__global__ void __launch_bounds__(256)
k_wmma_gemm_lds(const float* __restrict__ A, const _Float16* __restrict__ WT,
                const float* __restrict__ bias, float* __restrict__ C,
                int M, int K, int Nout, int act, int ngroups) {
  __shared__ __attribute__((aligned(32))) _Float16 lb[64 * 128];  // 16 KB max
  int ng = blockIdx.x % ngroups;       // 64-column group
  int mb = blockIdx.x / ngroups;       // 128-row block
  // ---- cooperative stage: contiguous 64*K halfs of WT -> LDS ----
  const _Float16* gsrc = WT + (size_t)ng * 64 * K;
  int chunks = (64 * K) >> 3;          // 16B chunks
  for (int idx = threadIdx.x; idx < chunks; idx += 256) {
#if HAVE_ASYNC_LDS
    // generic->int->addrspace ptr: LDS generic addr carries offset in addr[31:0]
    __builtin_amdgcn_global_load_async_to_lds_b128(
        (g_v4i*)(unsigned long long)(gsrc + 8 * idx),
        (l_v4i*)(unsigned)(unsigned long long)(lb + 8 * idx), 0, 0);
#else
    ((float4*)lb)[idx] = ((const float4*)gsrc)[idx];
#endif
  }
#if HAVE_ASYNC_LDS
  __builtin_amdgcn_s_wait_asynccnt(0);
#endif
  __syncthreads();

  int w     = threadIdx.x >> 5;
  int lane  = threadIdx.x & 31;
  int mt    = mb * 8 + w;
  if (mt >= (M >> 4)) return;          // after the only barrier
  int r     = lane & 15;
  int khalf = lane >> 4;               // 0: lanes 0-15, 1: lanes 16-31
  const float* arow = A + (size_t)(mt * 16 + r) * K;
  int nbase = ng * 64 + r;
  __builtin_prefetch(C + (size_t)(mt * 16) * Nout + ng * 64, 1, 0);  // global_prefetch
  v8f acc[4];
#pragma unroll
  for (int t = 0; t < 4; ++t) acc[t] = {};
#pragma unroll 4
  for (int k0 = 0; k0 < K; k0 += 32) {
    // A 16x32 f16 fragment: halfs 0..15 -> k0+khalf*16+j
    const float* ap = arow + k0 + khalf * 16;
    float4 a0 = ((const float4*)ap)[0];
    float4 a1 = ((const float4*)ap)[1];
    float4 a2 = ((const float4*)(ap + 8))[0];
    float4 a3 = ((const float4*)(ap + 8))[1];
    v16h a;
    a[0]  = (_Float16)a0.x; a[1]  = (_Float16)a0.y; a[2]  = (_Float16)a0.z; a[3]  = (_Float16)a0.w;
    a[4]  = (_Float16)a1.x; a[5]  = (_Float16)a1.y; a[6]  = (_Float16)a1.z; a[7]  = (_Float16)a1.w;
    a[8]  = (_Float16)a2.x; a[9]  = (_Float16)a2.y; a[10] = (_Float16)a2.z; a[11] = (_Float16)a2.w;
    a[12] = (_Float16)a3.x; a[13] = (_Float16)a3.y; a[14] = (_Float16)a3.z; a[15] = (_Float16)a3.w;
#pragma unroll
    for (int t = 0; t < 4; ++t) {
      // B 32x16 fragment from LDS: lanes 0-15 K=k0..k0+15, lanes 16-31 K=k0+16..k0+31
      const _Float16* lbp = lb + (size_t)(r + 16 * t) * K + k0 + khalf * 16;
      v16h b = *(const v16h*)lbp;
      acc[t] = __builtin_amdgcn_wmma_f32_16x16x32_f16(false, a, false, b, (short)0,
                                                      acc[t], false, false);
    }
  }
#pragma unroll
  for (int t = 0; t < 4; ++t) {
    int n = nbase + 16 * t;
    float bval = bias ? bias[n] : 0.0f;
#pragma unroll
    for (int i = 0; i < 8; ++i) {
      int mm = mt * 16 + khalf * 8 + i;  // C layout: VGPR i -> row i (+8 for upper lanes)
      float v = acc[t][i] + bval;
      if (act) v = fmaxf(v, 0.0f);
      C[(size_t)mm * Nout + n] = v;
    }
  }
}

// ---- small GEMM (Nout%16==0 only), pure-global path; used for Nout=16 ----
__global__ void k_wmma_gemm_small(const float* __restrict__ A, const _Float16* __restrict__ WT,
                                  const float* __restrict__ bias, float* __restrict__ C,
                                  int M, int K, int Nout, int act, int ntile_n, int total_tiles) {
  int wave = blockIdx.x * (blockDim.x >> 5) + (threadIdx.x >> 5);
  if (wave >= total_tiles) return;
  int lane  = threadIdx.x & 31;
  int nt    = wave % ntile_n;
  int mt    = wave / ntile_n;
  int r     = lane & 15;
  int khalf = lane >> 4;
  int m     = mt * 16 + r;
  int n     = nt * 16 + r;
  const float* arow = A + (size_t)m * K;
  v8f acc = {};
#pragma unroll 4
  for (int k0 = 0; k0 < K; k0 += 32) {
    const float* ap = arow + k0 + khalf * 16;
    float4 a0 = ((const float4*)ap)[0];
    float4 a1 = ((const float4*)ap)[1];
    float4 a2 = ((const float4*)(ap + 8))[0];
    float4 a3 = ((const float4*)(ap + 8))[1];
    v16h a;
    a[0]  = (_Float16)a0.x; a[1]  = (_Float16)a0.y; a[2]  = (_Float16)a0.z; a[3]  = (_Float16)a0.w;
    a[4]  = (_Float16)a1.x; a[5]  = (_Float16)a1.y; a[6]  = (_Float16)a1.z; a[7]  = (_Float16)a1.w;
    a[8]  = (_Float16)a2.x; a[9]  = (_Float16)a2.y; a[10] = (_Float16)a2.z; a[11] = (_Float16)a2.w;
    a[12] = (_Float16)a3.x; a[13] = (_Float16)a3.y; a[14] = (_Float16)a3.z; a[15] = (_Float16)a3.w;
    const _Float16* brow = WT + (size_t)n * K + k0 + khalf * 16;
    v16h b = *(const v16h*)brow;
    acc = __builtin_amdgcn_wmma_f32_16x16x32_f16(false, a, false, b, (short)0, acc,
                                                 false, false);
  }
  float bval = bias ? bias[n] : 0.0f;
#pragma unroll
  for (int i = 0; i < 8; ++i) {
    int mm = mt * 16 + khalf * 8 + i;
    float v = acc[i] + bval;
    if (act) v = fmaxf(v, 0.0f);
    C[(size_t)mm * Nout + n] = v;
  }
}

// ================= packing / conversion kernels =================
__global__ void k_cvt_wT(const float* __restrict__ W, _Float16* __restrict__ WT,
                         int K, int Nout, int Kpad, int Npad) {
  int t = blockIdx.x * blockDim.x + threadIdx.x;
  if (t >= Kpad * Npad) return;
  int k = t % Kpad, n = t / Kpad;
  float v = (k < K && n < Nout) ? W[(size_t)k * Nout + n] : 0.0f;
  WT[(size_t)n * Kpad + k] = (_Float16)v;
}

__global__ void k_pad_x(const float* __restrict__ x, float* __restrict__ xp) {
  int t = blockIdx.x * blockDim.x + threadIdx.x;
  if (t >= NN * 64) return;
  int n = t >> 6, k = t & 63;
  xp[t] = (k < FIN) ? x[(size_t)n * FIN + k] : 0.0f;
}

__global__ void k_pad_rows(const float* __restrict__ src, float* __restrict__ dst,
                           int rows, int rowspad) {
  int t = blockIdx.x * blockDim.x + threadIdx.x;
  if (t >= rowspad * DM) return;
  dst[t] = ((t / DM) < rows) ? src[t] : 0.0f;
}

__global__ void k_extract_out(const float* __restrict__ buf, const float* __restrict__ ob,
                              float* __restrict__ out) {
  int t = blockIdx.x * blockDim.x + threadIdx.x;
  if (t >= BG) return;
  out[t] = buf[(size_t)t * 16] + ob[0];
}

__global__ void k_fill(float* __restrict__ p, float v, int n) {
  int t = blockIdx.x * blockDim.x + threadIdx.x;
  if (t < n) p[t] = v;
}

// ================= GAT edge pipeline =================
DEVFN void edge_sd(int t, const int* __restrict__ src, const int* __restrict__ dst,
                   int& s, int& d) {
  if (t < EG) { s = src[t]; d = dst[t]; } else { s = t - EG; d = t - EG; }
}
DEVFN unsigned ordf(float f) {
  unsigned u = __float_as_uint(f);
  return (u & 0x80000000u) ? ~u : (u | 0x80000000u);
}
DEVFN float deordf(unsigned u) {
  return (u & 0x80000000u) ? __uint_as_float(u & 0x7fffffffu) : __uint_as_float(~u);
}

__global__ void k_gat_scores(const float* __restrict__ hp, const float* __restrict__ as_,
                             const float* __restrict__ ad_, float* __restrict__ es,
                             float* __restrict__ ed) {
  int t = blockIdx.x * blockDim.x + threadIdx.x;
  if (t >= NN * HG) return;
  int nidx = t >> 3, h = t & 7;
  const float* row = hp + (size_t)nidx * (HG * DM) + (size_t)h * DM;
  const float* sv  = as_ + h * DM;
  const float* dv  = ad_ + h * DM;
  float accs = 0.f, accd = 0.f;
  for (int c = 0; c < DM; c += 4) {
    float4 v = *(const float4*)(row + c);
    float4 a = *(const float4*)(sv + c);
    float4 b = *(const float4*)(dv + c);
    accs += v.x * a.x + v.y * a.y + v.z * a.z + v.w * a.w;
    accd += v.x * b.x + v.y * b.y + v.z * b.z + v.w * b.w;
  }
  es[t] = accs; ed[t] = accd;
}

__global__ void k_edge_p1(const int* __restrict__ src, const int* __restrict__ dst,
                          const float* __restrict__ es, const float* __restrict__ ed,
                          float* __restrict__ aedge, unsigned* __restrict__ amax) {
  int t = blockIdx.x * blockDim.x + threadIdx.x;
  if (t >= ET) return;
  int s, d; edge_sd(t, src, dst, s, d);
#pragma unroll
  for (int h = 0; h < HG; ++h) {
    float a = es[s * HG + h] + ed[d * HG + h];
    a = (a > 0.f) ? a : 0.2f * a;
    aedge[(size_t)t * HG + h] = a;
    atomicMax(&amax[d * HG + h], ordf(a));
  }
}

__global__ void k_edge_p2(const int* __restrict__ src, const int* __restrict__ dst,
                          float* __restrict__ aedge, const unsigned* __restrict__ amax,
                          float* __restrict__ den) {
  int t = blockIdx.x * blockDim.x + threadIdx.x;
  if (t >= ET) return;
  int s, d; edge_sd(t, src, dst, s, d); (void)s;
#pragma unroll
  for (int h = 0; h < HG; ++h) {
    float a  = aedge[(size_t)t * HG + h];
    float mx = deordf(amax[d * HG + h]);
    float e  = __expf(a - mx);
    aedge[(size_t)t * HG + h] = e;
    atomicAdd(&den[d * HG + h], e);
  }
}

__global__ void k_edge_p3(const int* __restrict__ src, const int* __restrict__ dst,
                          const float* __restrict__ hp, const float* __restrict__ aedge,
                          const float* __restrict__ den, float* __restrict__ acc) {
  int e = blockIdx.x * (blockDim.x >> 5) + (threadIdx.x >> 5);
  if (e >= ET) return;
  int lane = threadIdx.x & 31;
  int s, d; edge_sd(e, src, dst, s, d);
  float w[HG];
#pragma unroll
  for (int h = 0; h < HG; ++h)
    w[h] = aedge[(size_t)e * HG + h] / den[d * HG + h];
  int c0 = lane * 4;
  float4 v = {0.f, 0.f, 0.f, 0.f};
#pragma unroll
  for (int h = 0; h < HG; ++h) {
    float4 hv = *(const float4*)(hp + (size_t)s * (HG * DM) + (size_t)h * DM + c0);
    v.x += hv.x * w[h]; v.y += hv.y * w[h]; v.z += hv.z * w[h]; v.w += hv.w * w[h];
  }
  float* dp = acc + (size_t)d * DM + c0;
  atomicAdd(dp + 0, v.x * 0.125f);
  atomicAdd(dp + 1, v.y * 0.125f);
  atomicAdd(dp + 2, v.z * 0.125f);
  atomicAdd(dp + 3, v.w * 0.125f);
}

__global__ void k_bias_relu(float* __restrict__ x, const float* __restrict__ b, int rows) {
  int t = blockIdx.x * blockDim.x + threadIdx.x;
  if (t >= rows * DM) return;
  x[t] = fmaxf(x[t] + b[t & (DM - 1)], 0.f);
}

__global__ void k_add_bias(float* __restrict__ x, const float* __restrict__ b, int rows) {
  int t = blockIdx.x * blockDim.x + threadIdx.x;
  if (t >= rows * DM) return;
  x[t] += b[t & (DM - 1)];
}

__global__ void k_deg(const int* __restrict__ src, const int* __restrict__ dst,
                      float* __restrict__ deg) {
  int t = blockIdx.x * blockDim.x + threadIdx.x;
  if (t >= ET) return;
  int s, d; edge_sd(t, src, dst, s, d); (void)s;
  atomicAdd(&deg[d], 1.0f);
}

__global__ void k_gcn_scatter(const int* __restrict__ src, const int* __restrict__ dst,
                              const float* __restrict__ deg,
                              const float* __restrict__ zk, const float* __restrict__ zv,
                              float* __restrict__ Kacc, float* __restrict__ Vacc) {
  int e = blockIdx.x * (blockDim.x >> 5) + (threadIdx.x >> 5);
  if (e >= ET) return;
  int lane = threadIdx.x & 31;
  int s, d; edge_sd(e, src, dst, s, d);
  float norm = rsqrtf(deg[s]) * rsqrtf(deg[d]);
  int c0 = lane * 4;
  float4 kv = *(const float4*)(zk + (size_t)s * DM + c0);
  float4 vv = *(const float4*)(zv + (size_t)s * DM + c0);
  float* kp = Kacc + (size_t)d * DM + c0;
  float* vp = Vacc + (size_t)d * DM + c0;
  atomicAdd(kp + 0, kv.x * norm); atomicAdd(kp + 1, kv.y * norm);
  atomicAdd(kp + 2, kv.z * norm); atomicAdd(kp + 3, kv.w * norm);
  atomicAdd(vp + 0, vv.x * norm); atomicAdd(vp + 1, vv.y * norm);
  atomicAdd(vp + 2, vv.z * norm); atomicAdd(vp + 3, vv.w * norm);
}

// MAB attention with fused residual: Out[b,q,:] = Qp[b?,q,:] + softmax(QK^T/sqrt(128)) V
__global__ void k_attn(const float* __restrict__ Qp, int q_stride_b,
                       const float* __restrict__ Km, const float* __restrict__ Vm,
                       float* __restrict__ Out, int nq, int nk) {
  int b = blockIdx.x, h = blockIdx.y;
  int q = threadIdx.x;
  if (q >= nq) return;
  const float* qrow = Qp + (size_t)b * q_stride_b + (size_t)q * DM + h * DH;
  float qv[DH];
#pragma unroll
  for (int x = 0; x < DH; ++x) qv[x] = qrow[x];
  float mrun = -3.0e38f, lrun = 0.f;
  float av[DH];
#pragma unroll
  for (int x = 0; x < DH; ++x) av[x] = 0.f;
  const float scale = 0.08838834764831845f;   // 1/sqrt(128)
  for (int k = 0; k < nk; ++k) {
    const float* krow = Km + ((size_t)b * nk + k) * DM + h * DH;
    float sc = 0.f;
#pragma unroll
    for (int x = 0; x < DH; ++x) sc += qv[x] * krow[x];
    sc *= scale;
    float mn = fmaxf(mrun, sc);
    float rs = __expf(mrun - mn);
    float p  = __expf(sc - mn);
    lrun = lrun * rs + p;
    const float* vrow = Vm + ((size_t)b * nk + k) * DM + h * DH;
#pragma unroll
    for (int x = 0; x < DH; ++x) av[x] = av[x] * rs + p * vrow[x];
    mrun = mn;
  }
  float* orow = Out + ((size_t)b * nq + q) * DM + h * DH;
  float inv = 1.0f / lrun;
#pragma unroll
  for (int x = 0; x < DH; ++x) orow[x] = qrow[x] + av[x] * inv;
}

__global__ void k_resid_relu(const float* __restrict__ outp, const float* __restrict__ T,
                             float* __restrict__ dst, int n) {
  int t = blockIdx.x * blockDim.x + threadIdx.x;
  if (t >= n) return;
  dst[t] = outp[t] + fmaxf(T[t], 0.f);
}

// ---------------- workspace arena (float element offsets) ----------------
static constexpr size_t WS_A    = 0;                              // 33,554,432 : hp / GMT pool
static constexpr size_t WS_HB   = WS_A + (size_t)NN * 1024;
static constexpr size_t WS_HC   = WS_HB + (size_t)NN * DM;
static constexpr size_t WS_AEDG = WS_HC + (size_t)NN * DM;
static constexpr size_t WS_ES   = WS_AEDG + (size_t)ET * HG;
static constexpr size_t WS_ED   = WS_ES + (size_t)NN * HG;
static constexpr size_t WS_AMAX = WS_ED + (size_t)NN * HG;
static constexpr size_t WS_DEN  = WS_AMAX + (size_t)NN * HG;
static constexpr size_t WS_DEG  = WS_DEN + (size_t)NN * HG;
static constexpr size_t WS_XPAD = WS_DEG + (size_t)NN;
static constexpr size_t WS_S1P  = WS_XPAD + (size_t)NN * 64;
static constexpr size_t WS_QP1  = WS_S1P + 80 * DM;
static constexpr size_t WS_S3P  = WS_QP1 + 80 * DM;
static constexpr size_t WS_QP3  = WS_S3P + 16 * DM;
static constexpr size_t WS_OUTB = WS_QP3 + 16 * DM;
static constexpr size_t WS_WT   = WS_OUTB + (size_t)BG * 16;
// f16 weight arena offsets (in halfs)
static constexpr size_t WT_GAT0 = 0;
static constexpr size_t WT_GAT1 = 65536;
static constexpr size_t WT_GAT2 = 196608;
static constexpr size_t WT_LIN1 = 327680;
static constexpr size_t WT_LIN2 = 344064;
static constexpr size_t WT_FC   = 360448;
static constexpr size_t WT_OUT  = 376832;
static constexpr size_t WT_P1Q  = 378880;
static constexpr size_t WT_P1K  = 395264;
static constexpr size_t WT_P1V  = 411648;
static constexpr size_t WT_P1O  = 428032;
static constexpr size_t WT_P2Q  = 444416;
static constexpr size_t WT_P2K  = 460800;
static constexpr size_t WT_P2V  = 477184;
static constexpr size_t WT_P2O  = 493568;
static constexpr size_t WT_P3Q  = 509952;
static constexpr size_t WT_P3K  = 526336;
static constexpr size_t WT_P3V  = 542720;
static constexpr size_t WT_P3O  = 559104;
// GMT tensors inside the A pool (liveness-checked reuse)
static constexpr size_t G_Z    = WS_A;
static constexpr size_t G_ZWK  = WS_A + 4194304;
static constexpr size_t G_ZWV  = WS_A + 8388608;
static constexpr size_t G_KG   = WS_A + 12582912;
static constexpr size_t G_VG   = WS_A + 16777216;
static constexpr size_t G_OUT1 = WS_A + 20971520;
static constexpr size_t G_T1   = WS_A + 25886720;
static constexpr size_t G_H1   = WS_A + 0;
static constexpr size_t G_K2   = WS_A + 4915200;
static constexpr size_t G_V2   = WS_A + 9830400;
static constexpr size_t G_QP2  = WS_A + 14745600;
static constexpr size_t G_H2   = WS_A + 0;
static constexpr size_t G_K3   = WS_A + 4915200;
static constexpr size_t G_V3   = WS_A + 9830400;
static constexpr size_t G_OUT3 = WS_A + 14745600;
static constexpr size_t G_T3   = G_OUT3 + 65536;
static constexpr size_t G_H3   = G_T3 + 65536;
static constexpr size_t G_G1   = G_H3 + 65536;
static constexpr size_t G_G2   = G_G1 + 65536;

static inline int cdiv(int a, int b) { return (a + b - 1) / b; }

extern "C" void kernel_launch(void* const* d_in, const int* in_sizes, int n_in,
                              void* d_out, int out_size, void* d_ws, size_t ws_size,
                              hipStream_t stream) {
  (void)in_sizes; (void)out_size; (void)ws_size;
  const float* x   = (const float*)d_in[0];
  const int*   ei  = (const int*)d_in[1];
  const int*   src = ei;
  const int*   dst = ei + EG;
  const float* P[50];
  for (int i = 0; i < 50 && i < n_in; ++i) P[i] = (const float*)d_in[i];

  float*     ws = (float*)d_ws;
  _Float16*  wt = (_Float16*)(ws + WS_WT);

  auto fill = [&](float* p, float v, int n) {
    k_fill<<<cdiv(n, 256), 256, 0, stream>>>(p, v, n);
  };
  auto cvtw = [&](const float* W, size_t off, int K, int Nout, int Kpad, int Npad) {
    int n = Kpad * Npad;
    k_cvt_wT<<<cdiv(n, 256), 256, 0, stream>>>(W, wt + off, K, Nout, Kpad, Npad);
  };
  auto gemm = [&](const float* A, size_t wtoff, const float* bias, float* C,
                  int M, int Kp, int Nout, int act) {
    if (Nout % 64 == 0) {
      int ngroups = Nout / 64;
      int mblocks = cdiv(M / 16, 8);
      k_wmma_gemm_lds<<<mblocks * ngroups, 256, 0, stream>>>(A, wt + wtoff, bias, C,
                                                             M, Kp, Nout, act, ngroups);
    } else {
      int ntg = Nout / 16, tiles = (M / 16) * ntg;
      k_wmma_gemm_small<<<cdiv(tiles, 8), 256, 0, stream>>>(A, wt + wtoff, bias, C,
                                                            M, Kp, Nout, act, ntg, tiles);
    }
  };
  auto attn = [&](const float* Qp, int qstride, const float* Km, const float* Vm,
                  float* Out, int nq, int nk) {
    int bd = ((nq + 31) / 32) * 32;
    k_attn<<<dim3(BG, HT), bd, 0, stream>>>(Qp, qstride, Km, Vm, Out, nq, nk);
  };

  // ---- one-time-per-launch packing: pad x / seeds, transpose+f16 all weights ----
  k_pad_x<<<cdiv(NN * 64, 256), 256, 0, stream>>>(x, ws + WS_XPAD);
  k_pad_rows<<<cdiv(80 * DM, 256), 256, 0, stream>>>(P[20], ws + WS_S1P, SEEDS, 80);
  k_pad_rows<<<cdiv(16 * DM, 256), 256, 0, stream>>>(P[21], ws + WS_S3P, 1, 16);
  cvtw(P[4],  WT_GAT0, FIN, 1024, 64, 1024);
  cvtw(P[8],  WT_GAT1, DM, 1024, DM, 1024);
  cvtw(P[12], WT_GAT2, DM, 1024, DM, 1024);
  cvtw(P[16], WT_LIN1, DM, DM, DM, DM);
  cvtw(P[18], WT_LIN2, DM, DM, DM, DM);
  cvtw(P[46], WT_FC,   DM, DM, DM, DM);
  cvtw(P[48], WT_OUT,  DM, 1, DM, 16);
  const size_t wtp[12] = { WT_P1Q, WT_P1K, WT_P1V, WT_P1O,
                           WT_P2Q, WT_P2K, WT_P2V, WT_P2O,
                           WT_P3Q, WT_P3K, WT_P3V, WT_P3O };
  const int    pwi[12] = { 22, 23, 24, 25, 30, 31, 32, 33, 38, 39, 40, 41 };
  for (int i = 0; i < 12; ++i) cvtw(P[pwi[i]], wtp[i], DM, DM, DM, DM);

  // ================= 3 GAT layers =================
  const float* h_in = ws + WS_XPAD;
  int kin = 64;
  size_t wtg[3] = { WT_GAT0, WT_GAT1, WT_GAT2 };
  float* hping[2] = { ws + WS_HB, ws + WS_HC };
  for (int layer = 0; layer < 3; ++layer) {
    const float* asp = P[5 + 4 * layer];
    const float* adp = P[6 + 4 * layer];
    const float* bp  = P[7 + 4 * layer];
    float* hp   = ws + WS_A;
    float* hout = hping[layer & 1];        // L0->HB, L1->HC, L2->HB

    gemm(h_in, wtg[layer], nullptr, hp, NN, kin, HG * DM, 0);         // hp = h @ W
    k_gat_scores<<<cdiv(NN * HG, 256), 256, 0, stream>>>(hp, asp, adp,
                                                         ws + WS_ES, ws + WS_ED);
    fill(ws + WS_AMAX, 0.0f, NN * HG);
    fill(ws + WS_DEN,  0.0f, NN * HG);
    fill(hout,         0.0f, NN * DM);
    k_edge_p1<<<cdiv(ET, 256), 256, 0, stream>>>(src, dst, ws + WS_ES, ws + WS_ED,
                                                 ws + WS_AEDG, (unsigned*)(ws + WS_AMAX));
    k_edge_p2<<<cdiv(ET, 256), 256, 0, stream>>>(src, dst, ws + WS_AEDG,
                                                 (const unsigned*)(ws + WS_AMAX),
                                                 ws + WS_DEN);
    k_edge_p3<<<cdiv(ET, 8), 256, 0, stream>>>(src, dst, hp, ws + WS_AEDG,
                                               ws + WS_DEN, hout);
    k_bias_relu<<<cdiv(NN * DM, 256), 256, 0, stream>>>(hout, bp, NN);
    h_in = hout;
    kin  = DM;
  }

  // ================= GraphMultisetTransformer =================
  gemm(h_in, WT_LIN1, P[17], ws + G_Z, NN, DM, DM, 0);          // z = h@lin1 + b
  gemm(ws + G_Z, WT_P1K, nullptr, ws + G_ZWK, NN, DM, DM, 0);
  gemm(ws + G_Z, WT_P1V, nullptr, ws + G_ZWV, NN, DM, DM, 0);
  fill(ws + WS_DEG, 0.0f, NN);
  fill(ws + G_KG,   0.0f, NN * DM);
  fill(ws + G_VG,   0.0f, NN * DM);
  k_deg<<<cdiv(ET, 256), 256, 0, stream>>>(src, dst, ws + WS_DEG);
  k_gcn_scatter<<<cdiv(ET, 8), 256, 0, stream>>>(src, dst, ws + WS_DEG,
                                                 ws + G_ZWK, ws + G_ZWV,
                                                 ws + G_KG, ws + G_VG);
  k_add_bias<<<cdiv(NN * DM, 256), 256, 0, stream>>>(ws + G_KG, P[27], NN);
  k_add_bias<<<cdiv(NN * DM, 256), 256, 0, stream>>>(ws + G_VG, P[28], NN);

  // p1: PMA (GMPool_G). Qp1 shared across graphs (S1 broadcast).
  gemm(ws + WS_S1P, WT_P1Q, P[26], ws + WS_QP1, 80, DM, DM, 0);
  attn(ws + WS_QP1, 0, ws + G_KG, ws + G_VG, ws + G_OUT1, SEEDS, NPER);
  gemm(ws + G_OUT1, WT_P1O, P[29], ws + G_T1, BG * SEEDS, DM, DM, 0);
  k_resid_relu<<<cdiv(BG * SEEDS * DM, 256), 256, 0, stream>>>(ws + G_OUT1, ws + G_T1,
                                                               ws + G_H1, BG * SEEDS * DM);
  // p2: SAB
  gemm(ws + G_H1, WT_P2K, P[35], ws + G_K2,  BG * SEEDS, DM, DM, 0);
  gemm(ws + G_H1, WT_P2V, P[36], ws + G_V2,  BG * SEEDS, DM, DM, 0);
  gemm(ws + G_H1, WT_P2Q, P[34], ws + G_QP2, BG * SEEDS, DM, DM, 0);
  attn(ws + G_QP2, SEEDS * DM, ws + G_K2, ws + G_V2, ws + G_OUT1, SEEDS, SEEDS);
  gemm(ws + G_OUT1, WT_P2O, P[37], ws + G_T1, BG * SEEDS, DM, DM, 0);
  k_resid_relu<<<cdiv(BG * SEEDS * DM, 256), 256, 0, stream>>>(ws + G_OUT1, ws + G_T1,
                                                               ws + G_H2, BG * SEEDS * DM);
  // p3: PMA (GMPool_I), single seed shared across graphs.
  gemm(ws + G_H2, WT_P3K, P[43], ws + G_K3, BG * SEEDS, DM, DM, 0);
  gemm(ws + G_H2, WT_P3V, P[44], ws + G_V3, BG * SEEDS, DM, DM, 0);
  gemm(ws + WS_S3P, WT_P3Q, P[42], ws + WS_QP3, 16, DM, DM, 0);
  attn(ws + WS_QP3, 0, ws + G_K3, ws + G_V3, ws + G_OUT3, 1, SEEDS);
  gemm(ws + G_OUT3, WT_P3O, P[45], ws + G_T3, BG, DM, DM, 0);
  k_resid_relu<<<cdiv(BG * DM, 256), 256, 0, stream>>>(ws + G_OUT3, ws + G_T3,
                                                       ws + G_H3, BG * DM);

  // ================= head =================
  gemm(ws + G_H3, WT_LIN2, P[19], ws + G_G1, BG, DM, DM, 0);
  gemm(ws + G_G1, WT_FC,   P[47], ws + G_G2, BG, DM, DM, 1);
  gemm(ws + G_G2, WT_OUT,  nullptr, ws + WS_OUTB, BG, DM, 16, 0);
  k_extract_out<<<cdiv(BG, 256), 256, 0, stream>>>(ws + WS_OUTB, P[49], (float*)d_out);
}
